// DGCNNModule_16458314678665
// MI455X (gfx1250) — matrix operand order repaired and checked by reading, not compiled, not run on hardware
//
#include <hip/hip_runtime.h>
#include <hip/hip_bf16.h>

// ---------------------------------------------------------------------------
// DGCNN forward for MI455X (gfx1250, wave32, WMMA 16x16x32 f16).
// Dense GEMM work (edge-conv 1x1 convs, 1024-wide MLP) runs on
// v_wmma_f32_16x16x32_f16. h5 weight tiles are staged into LDS with the
// async copy engine (global_load_async_to_lds_b128 + s_wait_asynccnt),
// double-buffered so WMMA only ever waits on short-latency LDS loads.
// KNN distance+top-k is a fused wave-per-point kernel (selection-bound).
// Entire working set is L2-resident (192MB).
// ---------------------------------------------------------------------------

#define BB 8
#define NN 2048
#define KK 20

typedef __attribute__((ext_vector_type(16))) _Float16 v16h;
typedef __attribute__((ext_vector_type(8)))  float    v8f;

union F16Frag { v16h v; uint4 q[2]; };

typedef __attribute__((address_space(3))) _Float16 lds_f16_t;

__device__ __constant__ float kBnInv = 0.99999500003749981f; // 1/sqrt(1+1e-5)
#define NEGBIG (-1.0e30f)

static __device__ __forceinline__ float lrelu(float v) {
    return v >= 0.0f ? v : 0.2f * v;
}

// ---------------------------------------------------------------------------
// 0) Faithful torch-style rearrange: x[b,n,c] = points_flat[b, c*N + n]
// ---------------------------------------------------------------------------
__global__ void rearrange_kernel(const float* __restrict__ pts, float* __restrict__ x) {
    int e = blockIdx.x * blockDim.x + threadIdx.x;
    if (e >= BB * NN * 3) return;
    int b = e / (NN * 3);
    int r = e % (NN * 3);
    int n = r / 3, c = r % 3;
    x[e] = pts[b * (NN * 3) + c * NN + n];
}

// ---------------------------------------------------------------------------
// 1) fp32 weight -> f16, with column padding (layer1 K padded 6 -> 32)
// ---------------------------------------------------------------------------
__global__ void cvt_pad_f16_kernel(const float* __restrict__ src, _Float16* __restrict__ dst,
                                   int rows, int cs, int cd) {
    int e = blockIdx.x * blockDim.x + threadIdx.x;
    if (e >= rows * cd) return;
    int r = e / cd, c = e % cd;
    dst[e] = (c < cs) ? (_Float16)src[r * cs + c] : (_Float16)0.0f;
}

// ---------------------------------------------------------------------------
// 2) per-point squared norm
// ---------------------------------------------------------------------------
__global__ void sq_kernel(const float* __restrict__ feat, float* __restrict__ sq, int C) {
    int p = blockIdx.x * blockDim.x + threadIdx.x;
    if (p >= BB * NN) return;
    const float* f = feat + (size_t)p * C;
    float s = 0.0f;
    for (int c = 0; c < C; ++c) s += f[c] * f[c];
    sq[p] = s;
}

// ---------------------------------------------------------------------------
// 3) fused pairwise-distance + top-20: one wave per point, pd row in LDS,
//    20 rounds of wave-wide argmax via shfl_xor (wave32).
// ---------------------------------------------------------------------------
__global__ __launch_bounds__(128)
void knn_kernel(const float* __restrict__ feat, const float* __restrict__ sq,
                int* __restrict__ idx, int C) {
    extern __shared__ float pd[]; // 4 waves * NN floats
    int wave = threadIdx.x >> 5, lane = threadIdx.x & 31;
    int gp = blockIdx.x * 4 + wave;
    int b = gp / NN, i = gp % NN;
    const float* fb = feat + (size_t)b * NN * C;
    const float* fi = fb + (size_t)i * C;
    float sqi = sq[b * NN + i];
    float* row = pd + wave * NN;

    for (int j = lane; j < NN; j += 32) {
        const float* fj = fb + (size_t)j * C;
        float dot = 0.0f;
        for (int c = 0; c < C; ++c) dot += fi[c] * fj[c];
        row[j] = 2.0f * dot - sqi - sq[b * NN + j];
    }
    __syncthreads();

    for (int t = 0; t < KK; ++t) {
        float best = NEGBIG; int bj = NN;
        for (int j = lane; j < NN; j += 32) {
            float v = row[j];
            if (v > best) { best = v; bj = j; }
        }
        for (int off = 16; off > 0; off >>= 1) {
            float ov = __shfl_xor(best, off, 32);
            int   oj = __shfl_xor(bj, off, 32);
            if (ov > best || (ov == best && oj < bj)) { best = ov; bj = oj; }
        }
        if (lane == 0) idx[((size_t)b * NN + i) * KK + t] = bj;
        if ((bj & 31) == lane) row[bj] = NEGBIG; // retire winner
    }
}

// ---------------------------------------------------------------------------
// 4) fused edge conv: gather -> edge matrix (f16, LDS) -> WMMA vs W (f16) ->
//    BN+LReLU -> max over k=20. P points/block, R=P*20 rows (multiple of 16).
//    Weight fragments (nt,kt) hoisted to registers once per n-tile; per
//    m-tile all LDS A-fragments load as one batch, then a back-to-back
//    accumulator-chained WMMA burst.
// ---------------------------------------------------------------------------
template<int CIN, int COUT, int P>
__global__ __launch_bounds__(256)
void edge_conv_kernel(const float* __restrict__ feat, const int* __restrict__ knn,
                      const _Float16* __restrict__ Wh, const float* __restrict__ g,
                      const float* __restrict__ bnb,
                      float* __restrict__ out, _Float16* __restrict__ cat, int cat_off) {
    constexpr int KDIM = (2 * CIN < 32) ? 32 : 2 * CIN;
    constexpr int R  = P * KK;     // 320 / 160 / 80 : multiples of 16
    constexpr int MT = R / 16;
    constexpr int NT = COUT / 16;
    constexpr int KT = KDIM / 32;

    extern __shared__ char smem[];
    _Float16* Ae = (_Float16*)smem;                         // R*KDIM f16
    float* Dst   = (float*)(smem + (size_t)R * KDIM * 2);   // R*16 f32

    int tid = threadIdx.x;
    int b   = blockIdx.x / (NN / P);
    int pt0 = (blockIdx.x % (NN / P)) * P;
    const float* fb = feat + (size_t)b * NN * CIN;

    // ---- stage edge matrix [(nbr-ctr) | ctr | zero-pad] in f16 ----
    for (int e = tid; e < R * KDIM; e += 256) {
        int row = e / KDIM, c = e % KDIM;
        int p = row / KK, k = row % KK;
        int i = pt0 + p;
        float v;
        if (c < CIN) {
            int j = knn[((size_t)b * NN + i) * KK + k];
            v = fb[(size_t)j * CIN + c] - fb[(size_t)i * CIN + c];
        } else if (c < 2 * CIN) {
            v = fb[(size_t)i * CIN + (c - CIN)];
        } else {
            v = 0.0f;
        }
        Ae[e] = (_Float16)v;
    }
    __syncthreads();

    int wave = tid >> 5, lane = tid & 31;
    int lh = lane >> 4, ll = lane & 15;

    for (int nt = 0; nt < NT; ++nt) {
        // hoist all weight fragments for this n-tile into registers
        const _Float16* bbase = Wh + (size_t)(nt * 16 + ll) * KDIM + lh * 16;
        F16Frag bw[KT];
#pragma unroll
        for (int kt = 0; kt < KT; ++kt) {
            bw[kt].q[0] = *(const uint4*)(bbase + kt * 32);
            bw[kt].q[1] = *(const uint4*)(bbase + kt * 32 + 8);
        }
        if (nt + 1 < NT) // pull next weight tile toward L0 while we compute
            __builtin_prefetch(Wh + (size_t)((nt + 1) * 16 + ll) * KDIM + lh * 16, 0, 3);

        for (int mt = wave; mt < MT; mt += 8) {
            const _Float16* arow = Ae + (size_t)(mt * 16 + ll) * KDIM + lh * 16;
            F16Frag fa[KT];
#pragma unroll
            for (int kt = 0; kt < KT; ++kt) {   // one batch of ds_load_b128s
                fa[kt].q[0] = *(const uint4*)(arow + kt * 32);
                fa[kt].q[1] = *(const uint4*)(arow + kt * 32 + 8);
            }
            v8f acc = {};
#pragma unroll
            for (int kt = 0; kt < KT; ++kt)     // back-to-back WMMA burst
                acc = __builtin_amdgcn_wmma_f32_16x16x32_f16(
                    false, fa[kt].v, false, bw[kt].v, (short)0, acc, false, false);

            // BN + LReLU; D layout: M = r + 8*(lane>=16), N = lane&15
            int n = nt * 16 + ll;
            float sc = kBnInv * g[n];
            float bi = bnb[n];
            int mrow = mt * 16 + 8 * lh;
#pragma unroll
            for (int r = 0; r < 8; ++r) {
                float v = lrelu(acc[r] * sc + bi);
                Dst[(mrow + r) * 16 + ll] = v;
            }
        }
        __syncthreads();
        // max over k=20 neighbors per point, for this 16-col tile
        if (tid < P * 16) {
            int p = tid >> 4, n = tid & 15;
            float mx = NEGBIG;
            for (int k = 0; k < KK; ++k) mx = fmaxf(mx, Dst[(p * KK + k) * 16 + n]);
            int oc = nt * 16 + n;
            size_t o = ((size_t)b * NN + pt0 + p);
            out[o * COUT + oc] = mx;
            cat[o * 512 + cat_off + oc] = (_Float16)mx;
        }
        __syncthreads();
    }
}

// ---------------------------------------------------------------------------
// 5) h5 = lrelu(bn(cat @ W5^T)).
//    16 A-fragments resident per wave. Weight tiles (16 rows x 512 halves =
//    16KB, shared by all 8 waves) are double-buffered in LDS via the async
//    copy engine: each thread issues 4 global_load_async_to_lds_b128 per
//    tile; s_wait_asynccnt 0x4 retires the older tile while the next tile's
//    loads stay in flight. WMMA chain then only waits on LDS.
//    Partials: [B, 16 blocks, 8 waves, 1024 cols].
// ---------------------------------------------------------------------------
__global__ __launch_bounds__(256)
void h5_kernel(const _Float16* __restrict__ cat, const _Float16* __restrict__ W5h,
               const float* __restrict__ g5, const float* __restrict__ b5,
               float* __restrict__ pmax, float* __restrict__ psum) {
    __shared__ _Float16 wtile[2][16 * 512]; // 2 x 16KB weight tiles

    int tid = threadIdx.x;
    int wave = tid >> 5, lane = tid & 31;
    int lh = lane >> 4, ll = lane & 15;
    int blk = blockIdx.x & 15;
    int b   = blockIdx.x >> 4;
    int row0 = blk * 128 + wave * 16;
    const _Float16* Abase = cat + ((size_t)b * NN + row0) * 512 + (size_t)ll * 512 + lh * 16;

    // async-issue one 16KB weight tile into LDS buffer `buf`:
    // 256 threads x 4 chunks x 16B
    auto issue_tile = [&](int nt, int buf) {
        const _Float16* src = W5h + (size_t)nt * (16 * 512) + (size_t)tid * 8;
        _Float16* dst = &wtile[buf][tid * 8];
#pragma unroll
        for (int it = 0; it < 4; ++it) {
            unsigned lds_addr = (unsigned)(size_t)(lds_f16_t*)(dst + it * 2048);
            const _Float16* ga = src + it * 2048;
            asm volatile("global_load_async_to_lds_b128 %0, %1, off"
                         :: "v"(lds_addr), "v"(ga)
                         : "memory");
        }
    };

    // preload all 16 A fragments (128 VGPRs), resident across the nt loop
    F16Frag fa[16];
#pragma unroll
    for (int kt = 0; kt < 16; ++kt) {
        fa[kt].q[0] = *(const uint4*)(Abase + kt * 32);
        fa[kt].q[1] = *(const uint4*)(Abase + kt * 32 + 8);
    }

    issue_tile(0, 0);

    for (int nt = 0; nt < 64; ++nt) {
        int buf = nt & 1;
        if (nt + 1 < 64) {
            issue_tile(nt + 1, buf ^ 1);
            // in-order completion: <=4 outstanding leaves only the next
            // tile's loads in flight, guarantees current tile landed
            asm volatile("s_wait_asynccnt 0x4" ::: "memory");
        } else {
            asm volatile("s_wait_asynccnt 0x0" ::: "memory");
        }
        __syncthreads();

        const _Float16* bt = &wtile[buf][ll * 512 + lh * 16];
        // double-buffered LDS B-fragments: short ds latency, no global waits
        F16Frag fw[2];
        fw[0].q[0] = *(const uint4*)(bt);
        fw[0].q[1] = *(const uint4*)(bt + 8);
        v8f acc = {};
#pragma unroll
        for (int kt = 0; kt < 16; ++kt) {
            if (kt + 1 < 16) {
                fw[(kt + 1) & 1].q[0] = *(const uint4*)(bt + (kt + 1) * 32);
                fw[(kt + 1) & 1].q[1] = *(const uint4*)(bt + (kt + 1) * 32 + 8);
            }
            acc = __builtin_amdgcn_wmma_f32_16x16x32_f16(
                false, fa[kt].v, false, fw[kt & 1].v, (short)0, acc, false, false);
        }

        int n = nt * 16 + ll;
        float sc = kBnInv * g5[n], bi = b5[n];
        float mx = NEGBIG, sm = 0.0f;
#pragma unroll
        for (int r = 0; r < 8; ++r) {
            float v = lrelu(acc[r] * sc + bi);
            mx = fmaxf(mx, v);
            sm += v;
        }
        // combine the two half-row groups (lane L and L+16 share column n)
        float omx = __shfl_xor(mx, 16, 32);
        float osm = __shfl_xor(sm, 16, 32);
        mx = fmaxf(mx, omx);
        sm += osm;
        if (lane < 16) {
            size_t o = (((size_t)b * 16 + blk) * 8 + wave) * 1024 + n;
            pmax[o] = mx;
            psum[o] = sm;
        }
        __syncthreads(); // buffer-reuse safety before next iteration's issue
    }
}

// ---------------------------------------------------------------------------
// 6) pool reduce: feat[b, 0:1024] = max, feat[b, 1024:2048] = mean
// ---------------------------------------------------------------------------
__global__ void pool_reduce_kernel(const float* __restrict__ pmax, const float* __restrict__ psum,
                                   float* __restrict__ feat) {
    int e = blockIdx.x * blockDim.x + threadIdx.x;
    if (e >= BB * 1024) return;
    int b = e / 1024, col = e % 1024;
    float mx = NEGBIG, sm = 0.0f;
    for (int p = 0; p < 128; ++p) {
        size_t o = ((size_t)b * 128 + p) * 1024 + col;
        mx = fmaxf(mx, pmax[o]);
        sm += psum[o];
    }
    feat[b * 2048 + col] = mx;
    feat[b * 2048 + 1024 + col] = sm / (float)NN;
}

// ---------------------------------------------------------------------------
// 7) FC head: 2048 -> 512 -> 256 -> 40 (B=8, one block per batch row)
// ---------------------------------------------------------------------------
__global__ __launch_bounds__(256)
void fc_head_kernel(const float* __restrict__ feat,
                    const float* __restrict__ Wl1, const float* __restrict__ g6, const float* __restrict__ b6,
                    const float* __restrict__ Wl2, const float* __restrict__ bl2,
                    const float* __restrict__ g7, const float* __restrict__ b7,
                    const float* __restrict__ Wl3, const float* __restrict__ bl3,
                    float* __restrict__ out) {
    __shared__ float sf[2048];
    __shared__ float h1[512];
    __shared__ float h2[256];
    int b = blockIdx.x, tid = threadIdx.x;
    for (int c = tid; c < 2048; c += 256) sf[c] = feat[b * 2048 + c];
    __syncthreads();
    for (int o = tid; o < 512; o += 256) {
        const float* w = Wl1 + (size_t)o * 2048;
        float d = 0.0f;
        for (int c = 0; c < 2048; ++c) d += w[c] * sf[c];
        h1[o] = lrelu(d * kBnInv * g6[o] + b6[o]);
    }
    __syncthreads();
    if (tid < 256) {
        const float* w = Wl2 + (size_t)tid * 512;
        float d = bl2[tid];
        for (int c = 0; c < 512; ++c) d += w[c] * h1[c];
        h2[tid] = lrelu(d * kBnInv * g7[tid] + b7[tid]);
    }
    __syncthreads();
    if (tid < 40) {
        const float* w = Wl3 + (size_t)tid * 256;
        float d = bl3[tid];
        for (int c = 0; c < 256; ++c) d += w[c] * h2[c];
        out[b * 40 + tid] = d;
    }
}

// ---------------------------------------------------------------------------
// launcher
// ---------------------------------------------------------------------------
extern "C" void kernel_launch(void* const* d_in, const int* in_sizes, int n_in,
                              void* d_out, int out_size, void* d_ws, size_t ws_size,
                              hipStream_t stream) {
    const float* points = (const float*)d_in[0];
    const float* W1 = (const float*)d_in[1];  const float* g1 = (const float*)d_in[2];  const float* b1 = (const float*)d_in[3];
    const float* W2 = (const float*)d_in[4];  const float* g2 = (const float*)d_in[5];  const float* b2 = (const float*)d_in[6];
    const float* W3 = (const float*)d_in[7];  const float* g3 = (const float*)d_in[8];  const float* b3 = (const float*)d_in[9];
    const float* W4 = (const float*)d_in[10]; const float* g4 = (const float*)d_in[11]; const float* b4 = (const float*)d_in[12];
    const float* W5 = (const float*)d_in[13]; const float* g5 = (const float*)d_in[14]; const float* b5 = (const float*)d_in[15];
    const float* Wl1 = (const float*)d_in[16]; const float* g6 = (const float*)d_in[17]; const float* b6 = (const float*)d_in[18];
    const float* Wl2 = (const float*)d_in[19]; const float* bl2 = (const float*)d_in[20];
    const float* g7 = (const float*)d_in[21]; const float* b7 = (const float*)d_in[22];
    const float* Wl3 = (const float*)d_in[23]; const float* bl3 = (const float*)d_in[24];
    float* outp = (float*)d_out;

    // ---- workspace carving (all 256B aligned) ----
    char* ws = (char*)d_ws;
    size_t off = 0;
    auto alloc = [&](size_t bytes) { char* p = ws + off; off = (off + bytes + 255) & ~(size_t)255; return p; };
    float*    x    = (float*)alloc((size_t)BB * NN * 3 * 4);
    float*    sqb  = (float*)alloc((size_t)BB * NN * 4);
    int*      idx  = (int*)  alloc((size_t)BB * NN * KK * 4);
    float*    x1   = (float*)alloc((size_t)BB * NN * 64 * 4);
    float*    x2   = (float*)alloc((size_t)BB * NN * 64 * 4);
    float*    x3   = (float*)alloc((size_t)BB * NN * 128 * 4);
    float*    x4   = (float*)alloc((size_t)BB * NN * 256 * 4);
    _Float16* cat  = (_Float16*)alloc((size_t)BB * NN * 512 * 2);
    _Float16* W1h  = (_Float16*)alloc((size_t)64 * 32 * 2);
    _Float16* W2h  = (_Float16*)alloc((size_t)64 * 128 * 2);
    _Float16* W3h  = (_Float16*)alloc((size_t)128 * 128 * 2);
    _Float16* W4h  = (_Float16*)alloc((size_t)256 * 256 * 2);
    _Float16* W5h  = (_Float16*)alloc((size_t)1024 * 512 * 2);
    float*    pmax = (float*)alloc((size_t)BB * 128 * 1024 * 4);
    float*    psum = (float*)alloc((size_t)BB * 128 * 1024 * 4);
    float*    feat = (float*)alloc((size_t)BB * 2048 * 4);
    (void)ws_size; (void)n_in; (void)in_sizes; (void)out_size;

    // ---- 0) rearrange + weight conversion ----
    rearrange_kernel<<<(BB * NN * 3 + 255) / 256, 256, 0, stream>>>(points, x);
    cvt_pad_f16_kernel<<<(64 * 32 + 255) / 256, 256, 0, stream>>>(W1, W1h, 64, 6, 32);
    cvt_pad_f16_kernel<<<(64 * 128 + 255) / 256, 256, 0, stream>>>(W2, W2h, 64, 128, 128);
    cvt_pad_f16_kernel<<<(128 * 128 + 255) / 256, 256, 0, stream>>>(W3, W3h, 128, 128, 128);
    cvt_pad_f16_kernel<<<(256 * 256 + 255) / 256, 256, 0, stream>>>(W4, W4h, 256, 256, 256);
    cvt_pad_f16_kernel<<<(1024 * 512 + 255) / 256, 256, 0, stream>>>(W5, W5h, 1024, 512, 512);

    const int knnGrid = BB * NN / 4;
    const size_t knnLds = (size_t)4 * NN * 4;

    // ---- layer 1: C=3 -> 64 (K padded to 32) ----
    sq_kernel<<<(BB * NN + 255) / 256, 256, 0, stream>>>(x, sqb, 3);
    knn_kernel<<<knnGrid, 128, knnLds, stream>>>(x, sqb, idx, 3);
    edge_conv_kernel<3, 64, 16><<<BB * (NN / 16), 256, 320 * 32 * 2 + 320 * 16 * 4, stream>>>(
        x, idx, W1h, g1, b1, x1, cat, 0);

    // ---- layer 2: 64 -> 64 ----
    sq_kernel<<<(BB * NN + 255) / 256, 256, 0, stream>>>(x1, sqb, 64);
    knn_kernel<<<knnGrid, 128, knnLds, stream>>>(x1, sqb, idx, 64);
    edge_conv_kernel<64, 64, 8><<<BB * (NN / 8), 256, 160 * 128 * 2 + 160 * 16 * 4, stream>>>(
        x1, idx, W2h, g2, b2, x2, cat, 64);

    // ---- layer 3: 64 -> 128 ----
    sq_kernel<<<(BB * NN + 255) / 256, 256, 0, stream>>>(x2, sqb, 64);
    knn_kernel<<<knnGrid, 128, knnLds, stream>>>(x2, sqb, idx, 64);
    edge_conv_kernel<64, 128, 8><<<BB * (NN / 8), 256, 160 * 128 * 2 + 160 * 16 * 4, stream>>>(
        x2, idx, W3h, g3, b3, x3, cat, 128);

    // ---- layer 4: 128 -> 256 ----
    sq_kernel<<<(BB * NN + 255) / 256, 256, 0, stream>>>(x3, sqb, 128);
    knn_kernel<<<knnGrid, 128, knnLds, stream>>>(x3, sqb, idx, 128);
    edge_conv_kernel<128, 256, 4><<<BB * (NN / 4), 256, 80 * 256 * 2 + 80 * 16 * 4, stream>>>(
        x3, idx, W4h, g4, b4, x4, cat, 256);

    // ---- h5 + pooling ----
    h5_kernel<<<BB * 16, 256, 0, stream>>>(cat, W5h, g5, b5, pmax, psum);
    pool_reduce_kernel<<<(BB * 1024 + 255) / 256, 256, 0, stream>>>(pmax, psum, feat);

    // ---- FC head ----
    fc_head_kernel<<<BB, 256, 0, stream>>>(feat, Wl1, g6, b6, Wl2, bl2, g7, b7, Wl3, bl3, outp);
}
